// Attention_27994596836196
// MI455X (gfx1250) — matrix operand order repaired and checked by reading, not compiled, not run on hardware
//
#include <hip/hip_runtime.h>
#include <hip/hip_bf16.h>

// ---------------------------------------------------------------------------
// GQA attention block for MI455X (gfx1250, wave32, WMMA).
// All matmuls via v_wmma_f32_16x16x32_bf16 (bf16 in, f32 accum).
// Tile staging uses GLOBAL_LOAD_ASYNC_TO_LDS_B128 (ASYNCcnt) where no
// transpose is needed, with double-buffered K-steps in the GEMM.
// ---------------------------------------------------------------------------

typedef __attribute__((ext_vector_type(16))) __bf16 bf16x16;
typedef __attribute__((ext_vector_type(8)))  float  floatx8;

union Frag {
    uint4   u4[2];   // 32 bytes = 16 bf16 per lane
    bf16x16 v;
};

constexpr int kB      = 2;
constexpr int kS      = 2048;
constexpr int kHID    = 4096;
constexpr int kHEADS  = 32;
constexpr int kGROUPS = 8;
constexpr int kHD     = 128;
constexpr int kKV     = kGROUPS * kHD;   // 1024
constexpr int kM      = kB * kS;         // 4096 rows of x

// --- CDNA5 async copy helpers (inline asm; portable across toolchains) -----
__device__ __forceinline__ unsigned lds_addr(const void* p) {
    // Generic pointers to LDS carry the byte offset in addr[31:0].
    return (unsigned)(size_t)p;
}
__device__ __forceinline__ void async_ld_b128(unsigned lds, const void* g) {
    asm volatile("global_load_async_to_lds_b128 %0, %1, off"
                 :: "v"(lds), "v"((unsigned long long)(size_t)g)
                 : "memory");
}
__device__ __forceinline__ void wait_async_le2() {
    asm volatile("s_wait_asynccnt 0x2" ::: "memory");
}
__device__ __forceinline__ void wait_async_0() {
    asm volatile("s_wait_asynccnt 0x0" ::: "memory");
}

// ---------------------------------------------------------------------------
// Kernel 1/4: bf16 GEMM  C[M,N] = A[M,K] @ W[K,N], all row-major bf16.
// Block tile 128x128x32; 8 waves, each wave owns 32(M) x 64(N) = 2x4 WMMA
// tiles. A tile staged via async-to-LDS (double buffered); W tile staged
// TRANSPOSED in LDS so B-fragments are contiguous b128 loads.
// ---------------------------------------------------------------------------
__global__ __launch_bounds__(256)
void gemm_bf16_wmma(const __hip_bfloat16* __restrict__ A,
                    const __hip_bfloat16* __restrict__ W,
                    __hip_bfloat16* __restrict__ C,
                    int M, int N, int K) {
    constexpr int BM = 128, BN = 128, BK = 32, PAD = 8;
    __shared__ __hip_bfloat16 As[2][BM][BK + PAD];   // 80B pitch, 16B aligned
    __shared__ __hip_bfloat16 Bs[2][BN][BK + PAD];   // transposed tile: [n][k]

    const int t     = threadIdx.x;
    const int lane  = t & 31;
    const int wid   = t >> 5;        // 0..7
    const int wm    = wid >> 1;      // 0..3  (M direction)
    const int wn    = wid & 1;       // 0..1  (N direction)
    const int lrow  = lane & 15;
    const int khalf = lane >> 4;
    const int m0    = blockIdx.y * BM;
    const int n0    = blockIdx.x * BN;

    // Stage one K-step's tiles into buffer bufi.
    auto stage = [&](int k0, int bufi) {
        // A tile: 128x32 bf16, 2 async b128 per thread, no VGPR round-trip.
        #pragma unroll
        for (int i = 0; i < 2; i++) {
            int idx = t + i * 256;
            int r = idx >> 2, c = idx & 3;
            async_ld_b128(lds_addr(&As[bufi][r][c * 8]),
                          A + (size_t)(m0 + r) * K + k0 + c * 8);
        }
        // W tile transposed: read [k][n] chunks, scatter to Bs[n][k].
        #pragma unroll
        for (int i = 0; i < 2; i++) {
            int idx = t + i * 256;
            int kr = idx >> 4, nc = idx & 15;
            uint4 wv = *(const uint4*)(W + (size_t)(k0 + kr) * N + n0 + nc * 8);
            const __hip_bfloat16* ws = (const __hip_bfloat16*)&wv;
            #pragma unroll
            for (int j = 0; j < 8; j++) Bs[bufi][nc * 8 + j][kr] = ws[j];
        }
    };

    floatx8 acc[2][4] = {};

    stage(0, 0);
    int buf = 0;
    for (int k0 = 0; k0 < K; k0 += BK) {
        const bool has_next = (k0 + BK) < K;
        if (has_next) stage(k0 + BK, buf ^ 1);
        // Async loads complete in order: allowing 2 outstanding leaves the
        // next tile's 2 ops in flight while guaranteeing the current tile.
        if (has_next) wait_async_le2();
        else          wait_async_0();
        __syncthreads();

        // Fragments per ISA 7.12.2 layouts.
        Frag a[2], b[4];
        #pragma unroll
        for (int mt = 0; mt < 2; mt++) {
            const int r = wm * 32 + mt * 16 + lrow;
            a[mt].u4[0] = *(const uint4*)&As[buf][r][khalf * 8];
            a[mt].u4[1] = *(const uint4*)&As[buf][r][16 + khalf * 8];
        }
        #pragma unroll
        for (int nt = 0; nt < 4; nt++) {
            const int ccol = wn * 64 + nt * 16 + lrow;
            b[nt].u4[0] = *(const uint4*)&Bs[buf][ccol][khalf * 8];
            b[nt].u4[1] = *(const uint4*)&Bs[buf][ccol][16 + khalf * 8];
        }
        #pragma unroll
        for (int mt = 0; mt < 2; mt++)
            #pragma unroll
            for (int nt = 0; nt < 4; nt++)
                acc[mt][nt] = __builtin_amdgcn_wmma_f32_16x16x32_bf16(
                    false, a[mt].v, false, b[nt].v, (short)0, acc[mt][nt],
                    false, false);
        __syncthreads();   // reads of `buf` done before it is restaged
        buf ^= 1;
    }

    // Epilogue: D layout -> row = half*8 + vgpr, col = lane&15.
    #pragma unroll
    for (int mt = 0; mt < 2; mt++)
        #pragma unroll
        for (int nt = 0; nt < 4; nt++)
            #pragma unroll
            for (int r = 0; r < 8; r++) {
                int row = m0 + wm * 32 + mt * 16 + khalf * 8 + r;
                int col = n0 + wn * 64 + nt * 16 + lrow;
                C[(size_t)row * N + col] = __float2bfloat16(acc[mt][nt][r]);
            }
}

// ---------------------------------------------------------------------------
// Kernel 2: RoPE, interleaved-pair convention (x[2i], x[2i+1]) per head.
// ---------------------------------------------------------------------------
__global__ __launch_bounds__(256)
void rope_kernel(__hip_bfloat16* __restrict__ q, int heads,
                 const float* __restrict__ cos_half,
                 const float* __restrict__ sin_half, int total) {
    int i = blockIdx.x * blockDim.x + threadIdx.x;
    if (i >= total) return;
    int p    = i & 63;            // pair index within head (HD/2 = 64)
    int rest = i >> 6;
    int h    = rest % heads;
    int row  = rest / heads;      // b*S + s
    int s    = row % kS;
    size_t base = (size_t)row * heads * kHD + h * kHD + 2 * p;
    float x1 = __bfloat162float(q[base]);
    float x2 = __bfloat162float(q[base + 1]);
    float c  = cos_half[s * 64 + p];
    float sn = sin_half[s * 64 + p];
    q[base]     = __float2bfloat16(x1 * c - x2 * sn);
    q[base + 1] = __float2bfloat16(x1 * sn + x2 * c);
}

// ---------------------------------------------------------------------------
// Kernel 3: causal flash attention. One WG = 128 query rows of one head.
// 8 waves x 16 query rows. Key blocks of 64. Online softmax in f32.
// Score WMMA: A = Q rows (from global), B = K tile (async-staged; its natural
// [key][d] layout IS the B^T layout for Q.K^T).
// PV WMMA:    A = P (via LDS layout fixup), B = V tile staged transposed.
// ---------------------------------------------------------------------------
__global__ __launch_bounds__(256)
void flash_attn_wmma(const __hip_bfloat16* __restrict__ Q,
                     const __hip_bfloat16* __restrict__ Kc,
                     const __hip_bfloat16* __restrict__ Vc,
                     __hip_bfloat16* __restrict__ O) {
    constexpr int PAD = 8;
    __shared__ __hip_bfloat16 Kt[64][kHD + PAD];    // [key][d]
    __shared__ __hip_bfloat16 Vt[kHD][64 + PAD];    // [d][key] (transposed)
    __shared__ __hip_bfloat16 Pt[128][64 + PAD];    // [qrow][key]

    const int t     = threadIdx.x;
    const int lane  = t & 31;
    const int w     = t >> 5;       // wave id: rows w*16 .. w*16+15 of the tile
    const int lrow  = lane & 15;
    const int khalf = lane >> 4;
    const int q0    = blockIdx.x * 128;
    const int h     = blockIdx.y;
    const int b     = blockIdx.z;
    const int g     = h >> 2;       // h / REP
    const float scale = 0.08838834764831845f;   // 1/sqrt(128)

    // Q fragments straight from global: per lane, row q0+w*16+lrow is contiguous.
    const __hip_bfloat16* qrowp =
        Q + ((size_t)(b * kS + q0 + w * 16 + lrow)) * kHID + h * kHD;
    Frag aq[4];
    #pragma unroll
    for (int ks = 0; ks < 4; ks++) {
        aq[ks].u4[0] = *(const uint4*)(qrowp + ks * 32 + khalf * 8);
        aq[ks].u4[1] = *(const uint4*)(qrowp + ks * 32 + 16 + khalf * 8);
    }

    floatx8 o[8] = {};
    float m_r[8], l_r[8];
    #pragma unroll
    for (int r = 0; r < 8; r++) { m_r[r] = -3.0e38f; l_r[r] = 0.f; }

    const int kb_end = (q0 >> 6) + 2;   // causal: keys <= q0+127
    for (int kb = 0; kb < kb_end; kb++) {
        __syncthreads();
        // K tile 64x128 (natural layout) via async copy to LDS.
        #pragma unroll
        for (int i = 0; i < 4; i++) {
            int idx = t + i * 256;
            int r = idx >> 4, c = idx & 15;
            async_ld_b128(lds_addr(&Kt[r][c * 8]),
                          Kc + ((size_t)(b * kS + kb * 64 + r)) * kKV +
                              g * kHD + c * 8);
        }
        // V tile transposed to Vt[d][key] (overlaps with async K in flight).
        #pragma unroll
        for (int i = 0; i < 4; i++) {
            int idx = t + i * 256;
            int kr = idx >> 4, c = idx & 15;
            uint4 vv = *(const uint4*)(Vc + ((size_t)(b * kS + kb * 64 + kr)) * kKV +
                                       g * kHD + c * 8);
            const __hip_bfloat16* vs = (const __hip_bfloat16*)&vv;
            #pragma unroll
            for (int j = 0; j < 8; j++) Vt[c * 8 + j][kr] = vs[j];
        }
        wait_async_0();
        __syncthreads();

        // Scores S = Q @ K^T : 4 k-steps (HD=128), 4 key tiles of 16.
        floatx8 sacc[4] = {};
        #pragma unroll
        for (int ks = 0; ks < 4; ks++) {
            #pragma unroll
            for (int nt = 0; nt < 4; nt++) {
                Frag bk;
                const int kcol = nt * 16 + lrow;
                bk.u4[0] = *(const uint4*)&Kt[kcol][ks * 32 + khalf * 8];
                bk.u4[1] = *(const uint4*)&Kt[kcol][ks * 32 + 16 + khalf * 8];
                sacc[nt] = __builtin_amdgcn_wmma_f32_16x16x32_bf16(
                    false, aq[ks].v, false, bk.v, (short)0, sacc[nt],
                    false, false);
            }
        }

        // Scale + causal mask (D layout: row = khalf*8+r, col = lane&15).
        #pragma unroll
        for (int nt = 0; nt < 4; nt++) {
            int key = kb * 64 + nt * 16 + lrow;
            #pragma unroll
            for (int r = 0; r < 8; r++) {
                int qrow = q0 + w * 16 + khalf * 8 + r;
                float v = sacc[nt][r] * scale;
                if (key > qrow) v = -1.0e30f;
                sacc[nt][r] = v;
            }
        }

        // Online softmax: row reductions live in 16-lane halves.
        #pragma unroll
        for (int r = 0; r < 8; r++) {
            float lm = fmaxf(fmaxf(sacc[0][r], sacc[1][r]),
                             fmaxf(sacc[2][r], sacc[3][r]));
            #pragma unroll
            for (int off = 8; off; off >>= 1)
                lm = fmaxf(lm, __shfl_xor(lm, off, 32));
            float mnew = fmaxf(m_r[r], lm);
            float corr = __expf(m_r[r] - mnew);
            m_r[r] = mnew;
            l_r[r] *= corr;
            #pragma unroll
            for (int d = 0; d < 8; d++) o[d][r] *= corr;
            float rowsum = 0.f;
            #pragma unroll
            for (int nt = 0; nt < 4; nt++) {
                float p = __expf(sacc[nt][r] - mnew);
                rowsum += p;
                Pt[w * 16 + khalf * 8 + r][nt * 16 + lrow] = __float2bfloat16(p);
            }
            #pragma unroll
            for (int off = 8; off; off >>= 1)
                rowsum += __shfl_xor(rowsum, off, 32);
            l_r[r] += rowsum;
        }
        __syncthreads();

        // O += P @ V : contraction over 64 keys (2 k-steps), 8 HD tiles of 16.
        #pragma unroll
        for (int ks = 0; ks < 2; ks++) {
            Frag ap;
            ap.u4[0] = *(const uint4*)&Pt[w * 16 + lrow][ks * 32 + khalf * 8];
            ap.u4[1] = *(const uint4*)&Pt[w * 16 + lrow][ks * 32 + 16 + khalf * 8];
            #pragma unroll
            for (int d = 0; d < 8; d++) {
                Frag bv;
                const int drow = d * 16 + lrow;
                bv.u4[0] = *(const uint4*)&Vt[drow][ks * 32 + khalf * 8];
                bv.u4[1] = *(const uint4*)&Vt[drow][ks * 32 + 16 + khalf * 8];
                o[d] = __builtin_amdgcn_wmma_f32_16x16x32_bf16(
                    false, ap.v, false, bv.v, (short)0, o[d], false, false);
            }
        }
    }

    // Normalize and write attention output in [b, s, h, d] = [B*S, HID] layout.
    #pragma unroll
    for (int d = 0; d < 8; d++)
        #pragma unroll
        for (int r = 0; r < 8; r++) {
            int qrow = q0 + w * 16 + khalf * 8 + r;
            int col  = h * kHD + d * 16 + lrow;
            float val = o[d][r] / l_r[r];
            O[((size_t)(b * kS + qrow)) * kHID + col] = __float2bfloat16(val);
        }
}

// ---------------------------------------------------------------------------
// Host-side orchestration.
// ---------------------------------------------------------------------------
extern "C" void kernel_launch(void* const* d_in, const int* in_sizes, int n_in,
                              void* d_out, int out_size, void* d_ws, size_t ws_size,
                              hipStream_t stream) {
    const __hip_bfloat16* x        = (const __hip_bfloat16*)d_in[0];
    const float*          cos_half = (const float*)d_in[1];
    const float*          sin_half = (const float*)d_in[2];
    const __hip_bfloat16* w_q      = (const __hip_bfloat16*)d_in[3];
    const __hip_bfloat16* w_k      = (const __hip_bfloat16*)d_in[4];
    const __hip_bfloat16* w_v      = (const __hip_bfloat16*)d_in[5];
    const __hip_bfloat16* w_o      = (const __hip_bfloat16*)d_in[6];
    __hip_bfloat16*       out      = (__hip_bfloat16*)d_out;

    // Workspace: Q (32MB) | K (8MB) | V (8MB) | attn_out (32MB) = 80MB.
    __hip_bfloat16* Q0 = (__hip_bfloat16*)d_ws;
    __hip_bfloat16* K0 = Q0 + (size_t)kM * kHID;
    __hip_bfloat16* V0 = K0 + (size_t)kM * kKV;
    __hip_bfloat16* AO = V0 + (size_t)kM * kKV;

    // 1) QKV projections.
    gemm_bf16_wmma<<<dim3(kHID / 128, kM / 128), 256, 0, stream>>>(
        x, w_q, Q0, kM, kHID, kHID);
    gemm_bf16_wmma<<<dim3(kKV / 128, kM / 128), 256, 0, stream>>>(
        x, w_k, K0, kM, kKV, kHID);
    gemm_bf16_wmma<<<dim3(kKV / 128, kM / 128), 256, 0, stream>>>(
        x, w_v, V0, kM, kKV, kHID);

    // 2) RoPE on Q and K (in place).
    int totQ = kB * kS * kHEADS * (kHD / 2);
    rope_kernel<<<totQ / 256, 256, 0, stream>>>(Q0, kHEADS, cos_half, sin_half, totQ);
    int totK = kB * kS * kGROUPS * (kHD / 2);
    rope_kernel<<<totK / 256, 256, 0, stream>>>(K0, kGROUPS, cos_half, sin_half, totK);

    // 3) Causal GQA flash attention.
    flash_attn_wmma<<<dim3(kS / 128, kHEADS, kB), 256, 0, stream>>>(Q0, K0, V0, AO);

    // 4) Output projection.
    gemm_bf16_wmma<<<dim3(kHID / 128, kM / 128), 256, 0, stream>>>(
        AO, w_o, out, kM, kHID, kHID);
}
